// CrfRnnLayerSPAT_66168266162188
// MI455X (gfx1250) — compile-verified, matrix operand order checked by reading, not run on hardware
//
#include <hip/hip_runtime.h>
#include <hip/hip_bf16.h>

// CRF-RNN (CrfRnnLayerSPAT) for MI455X / gfx1250.
// Memory-bound pipeline; the 21x21 per-pixel transform is done with
// v_wmma_f32_16x16x32_f16 (A = padded weight matrix, B = 16-pixel tile),
// fused with the attachment term and the q update.

#define NC 21
#define HW 512
#define NPIX (512 * 512)
#define NB4 256   // blocks for masked partial reduction

typedef __attribute__((ext_vector_type(16))) _Float16 v16h;
typedef __attribute__((ext_vector_type(8)))  float    v8f;

__device__ __forceinline__ void gauss_taps(float k[9]) {
  float s = 0.f;
#pragma unroll
  for (int j = 0; j < 9; ++j) {
    float t = (float)(j - 4) * (1.0f / 3.0f);
    k[j] = __expf(-0.5f * t * t);
    s += k[j];
  }
  float inv = 1.0f / s;
#pragma unroll
  for (int j = 0; j < 9; ++j) k[j] *= inv;
}

// ---- init: transpose unaries (H,W,C)->(C,H,W) into ut, and q = ut ----
__global__ __launch_bounds__(256) void k0_init(const float* __restrict__ un,
                                               float* __restrict__ ut,
                                               float* __restrict__ q) {
  int p = blockIdx.x * 256 + threadIdx.x;
#pragma unroll
  for (int c = 0; c < NC; ++c) {
    float v = un[p * NC + c];
    ut[c * NPIX + p] = v;
    q[c * NPIX + p]  = v;
  }
}

// ---- M = compat @ (Ws + Wb), f16, zero-padded to 32x32 ----
__global__ __launch_bounds__(1024) void km_mat(const float* __restrict__ wsk,
                                               const float* __restrict__ wbk,
                                               const float* __restrict__ comp,
                                               _Float16* __restrict__ Mh) {
  int t = threadIdx.x;
  int i = t >> 5, j = t & 31;
  float acc = 0.f;
  if (i < NC && j < NC) {
    for (int k = 0; k < NC; ++k)
      acc += comp[i * NC + k] * (wsk[k * NC + j] + wbk[k * NC + j]);
  }
  Mh[i * 32 + j] = (_Float16)acc;
}

// ---- softmax over channels per pixel ----
__global__ __launch_bounds__(256) void k1_softmax(const float* __restrict__ q,
                                                  float* __restrict__ sm) {
  int p = blockIdx.x * 256 + threadIdx.x;
  float v[NC];
  float mx = -3.402823466e38f;
#pragma unroll
  for (int c = 0; c < NC; ++c) { v[c] = q[c * NPIX + p]; mx = fmaxf(mx, v[c]); }
  float s = 0.f;
#pragma unroll
  for (int c = 0; c < NC; ++c) { v[c] = __expf(v[c] - mx); s += v[c]; }
  float inv = 1.0f / s;
#pragma unroll
  for (int c = 0; c < NC; ++c) sm[c * NPIX + p] = v[c] * inv;
}

// ---- separable blur, vertical pass (zero padding, SAME) ----
__global__ __launch_bounds__(256) void k2_blur_y(const float* __restrict__ in,
                                                 float* __restrict__ out) {
  int p = blockIdx.x * 256 + threadIdx.x;
  int c = blockIdx.y;
  int y = p >> 9, x = p & 511;
  float k[9]; gauss_taps(k);
  const float* pl = in + (size_t)c * NPIX;
  float acc = 0.f;
#pragma unroll
  for (int j = 0; j < 9; ++j) {
    int yy = y + j - 4;
    if ((unsigned)yy < (unsigned)HW) acc += k[j] * pl[yy * HW + x];
  }
  out[(size_t)c * NPIX + p] = acc;
}

// ---- horizontal pass + analytic norm (blur-of-ones = f(y)*f(x)) ----
__global__ __launch_bounds__(256) void k3_blur_x(const float* __restrict__ in,
                                                 float* __restrict__ out) {
  int p = blockIdx.x * 256 + threadIdx.x;
  int c = blockIdx.y;
  int y = p >> 9, x = p & 511;
  float k[9]; gauss_taps(k);
  const float* row = in + (size_t)c * NPIX + y * HW;
  float acc = 0.f, fx = 0.f, fy = 0.f;
#pragma unroll
  for (int j = 0; j < 9; ++j) {
    int xx = x + j - 4;
    if ((unsigned)xx < (unsigned)HW) { acc += k[j] * row[xx]; fx += k[j]; }
    int yy = y + j - 4;
    if ((unsigned)yy < (unsigned)HW) fy += k[j];
  }
  out[(size_t)c * NPIX + p] = acc / (fx * fy);
}

// ---- masked exp-sum partial reduction (deterministic fixed-order) ----
// spm is used transposed in the reference: mask(y,x) = sp_map[x*W + y].
// We scan sp_map in its native order (coalesced) and transpose the pixel index.
__global__ __launch_bounds__(256) void k4a_partial(const float* __restrict__ sm,
                                                   const int* __restrict__ spmap,
                                                   const int* __restrict__ spi, int iter,
                                                   float* __restrict__ part) {
  __shared__ float red[256];
  int tid = threadIdx.x;
  int idx = spi[iter];
  float a1[NC], a2[NC];
#pragma unroll
  for (int c = 0; c < NC; ++c) { a1[c] = 0.f; a2[c] = 0.f; }
  float n1 = 0.f, n2 = 0.f;
  int s0 = blockIdx.x * (NPIX / NB4);
  for (int it = 0; it < (NPIX / NB4) / 256; ++it) {
    int s = s0 + it * 256 + tid;
    int spv = spmap[s];
    if (spv == idx || spv == idx + 1) {
      int x = s >> 9, y = s & 511;      // s = x*W + y (transposed map)
      int p = y * HW + x;
      if (spv == idx) {
        n1 += 1.f;
#pragma unroll
        for (int c = 0; c < NC; ++c) a1[c] += __expf(sm[c * NPIX + p]);
      } else {
        n2 += 1.f;
#pragma unroll
        for (int c = 0; c < NC; ++c) a2[c] += __expf(sm[c * NPIX + p]);
      }
    }
  }
  float* pb = part + blockIdx.x * 48;
#pragma unroll
  for (int c = 0; c < NC; ++c) {
    red[tid] = a1[c]; __syncthreads();
    for (int st = 128; st; st >>= 1) { if (tid < st) red[tid] += red[tid + st]; __syncthreads(); }
    if (tid == 0) pb[c] = red[0];
    __syncthreads();
  }
#pragma unroll
  for (int c = 0; c < NC; ++c) {
    red[tid] = a2[c]; __syncthreads();
    for (int st = 128; st; st >>= 1) { if (tid < st) red[tid] += red[tid + st]; __syncthreads(); }
    if (tid == 0) pb[NC + c] = red[0];
    __syncthreads();
  }
  red[tid] = n1; __syncthreads();
  for (int st = 128; st; st >>= 1) { if (tid < st) red[tid] += red[tid + st]; __syncthreads(); }
  if (tid == 0) pb[42] = red[0];
  __syncthreads();
  red[tid] = n2; __syncthreads();
  for (int st = 128; st; st >>= 1) { if (tid < st) red[tid] += red[tid + st]; __syncthreads(); }
  if (tid == 0) pb[43] = red[0];
}

// ---- finish: B[c] = log((N - n) + sum_exp) == logsumexp(sm*mask over plane) ----
__global__ __launch_bounds__(64) void k4b_finish(const float* __restrict__ part,
                                                 float* __restrict__ B12) {
  int c = threadIdx.x;
  if (c < 2 * NC) {
    float s = 0.f, n = 0.f;
    int cnt = (c < NC) ? 42 : 43;
    for (int b = 0; b < NB4; ++b) { s += part[b * 48 + c]; n += part[b * 48 + cnt]; }
    B12[c] = logf((float)NPIX - n + s);
  }
}

// ---- WMMA fused update: q = u - (M @ S) - att ----
// A (16x32 f16): lane m = lane&15, k = 16*(j>>3) + 8*g + (j&7)
// B (32x16 f16): lane n = lane&15, same k striping
// C/D (16x16 f32): row m = v + 8*g, col n = lane&15
__global__ __launch_bounds__(256) void k5_update(
    const float* __restrict__ S, const float* __restrict__ sm,
    const float* __restrict__ ut, const _Float16* __restrict__ Mh,
    const float* __restrict__ B12, const int* __restrict__ spmap,
    const int* __restrict__ spi, int iter, const float* __restrict__ low_w,
    const float* __restrict__ high_w, float* __restrict__ q) {
  int lane = threadIdx.x & 31;
  int wave = (blockIdx.x * 256 + threadIdx.x) >> 5;
  int base = wave * 16;                 // 16 pixels per wave
  int ln15 = lane & 15;
  int g = lane >> 4;

  v16h bfrag;
#pragma unroll
  for (int j = 0; j < 16; ++j) {
    int k = 16 * (j >> 3) + 8 * g + (j & 7);
    float v = (k < NC) ? S[k * NPIX + base + ln15] : 0.0f;
    bfrag[j] = (_Float16)v;
  }
  v16h a0, a1;
#pragma unroll
  for (int j = 0; j < 16; ++j) {
    int k = 16 * (j >> 3) + 8 * g + (j & 7);
    a0[j] = Mh[ln15 * 32 + k];          // output rows 0..15
    a1[j] = Mh[(16 + ln15) * 32 + k];   // output rows 16..31 (16..20 valid)
  }
  v8f zero = {};
  v8f d0 = __builtin_amdgcn_wmma_f32_16x16x32_f16(false, a0, false, bfrag,
                                                  (short)0, zero, false, false);
  v8f d1 = __builtin_amdgcn_wmma_f32_16x16x32_f16(false, a1, false, bfrag,
                                                  (short)0, zero, false, false);

  int p = base + ln15;
  int y = p >> 9, x = p & 511;
  int spv = spmap[x * HW + y];          // transposed superpixel map
  int idx = spi[iter];
  float m1 = (spv == idx)     ? 1.f : 0.f;
  float m2 = (spv == idx + 1) ? 1.f : 0.f;
  float hw0 = high_w[0], hw1 = high_w[1];
#pragma unroll
  for (int v = 0; v < 8; ++v) {
    {
      int m = v + 8 * g;
      float sv = sm[m * NPIX + p];
      float qm = sv + ((sv == 0.f) ? 1.f : 0.f);
      float c1 = m1 * B12[m];
      float c2 = m2 * B12[NC + m];
      float ft_sp  = c1 / qm;
      float ft_att = (c1 + c2) / qm;
      float att = low_w[m] * ft_sp + hw0 * (1.f - ft_sp)
                + low_w[NC + m] * ft_att + hw1 * (1.f - ft_att);
      q[m * NPIX + p] = ut[m * NPIX + p] - d0[v] - att;
    }
    int mm = 16 + v + 8 * g;
    if (mm < NC) {
      float sv = sm[mm * NPIX + p];
      float qm = sv + ((sv == 0.f) ? 1.f : 0.f);
      float c1 = m1 * B12[mm];
      float c2 = m2 * B12[NC + mm];
      float ft_sp  = c1 / qm;
      float ft_att = (c1 + c2) / qm;
      float att = low_w[mm] * ft_sp + hw0 * (1.f - ft_sp)
                + low_w[NC + mm] * ft_att + hw1 * (1.f - ft_att);
      q[mm * NPIX + p] = ut[mm * NPIX + p] - d1[v] - att;
    }
  }
}

// ---- final transpose (C,H,W) -> (H,W,C) ----
__global__ __launch_bounds__(256) void k6_output(const float* __restrict__ q,
                                                 float* __restrict__ out) {
  int p = blockIdx.x * 256 + threadIdx.x;
#pragma unroll
  for (int c = 0; c < NC; ++c) out[p * NC + c] = q[c * NPIX + p];
}

extern "C" void kernel_launch(void* const* d_in, const int* in_sizes, int n_in,
                              void* d_out, int out_size, void* d_ws, size_t ws_size,
                              hipStream_t stream) {
  const float* un    = (const float*)d_in[0];
  // d_in[1] = rgb: unused (the reference's stand-in filter ignores it)
  const int*   spmap = (const int*)d_in[2];
  const int*   spi   = (const int*)d_in[3];
  const float* wsk   = (const float*)d_in[4];
  const float* wbk   = (const float*)d_in[5];
  const float* comp  = (const float*)d_in[6];
  const float* loww  = (const float*)d_in[7];
  const float* highw = (const float*)d_in[8];
  float* out = (float*)d_out;

  char* ws = (char*)d_ws;
  const size_t plane = (size_t)NC * NPIX * sizeof(float);  // ~22 MB
  float*    ut   = (float*)(ws + 0 * plane);
  float*    q    = (float*)(ws + 1 * plane);
  float*    sm   = (float*)(ws + 2 * plane);
  float*    tbuf = (float*)(ws + 3 * plane);
  float*    S    = (float*)(ws + 4 * plane);
  _Float16* Mh   = (_Float16*)(ws + 5 * plane);
  float*    part = (float*)(ws + 5 * plane + 4096);
  float*    B12  = (float*)(ws + 5 * plane + 4096 + (size_t)NB4 * 48 * sizeof(float));

  dim3 b256(256);
  k0_init<<<NPIX / 256, b256, 0, stream>>>(un, ut, q);
  km_mat<<<1, 1024, 0, stream>>>(wsk, wbk, comp, Mh);
  for (int i = 0; i < 5; ++i) {
    k1_softmax<<<NPIX / 256, b256, 0, stream>>>(q, sm);
    k2_blur_y<<<dim3(NPIX / 256, NC), b256, 0, stream>>>(sm, tbuf);
    k3_blur_x<<<dim3(NPIX / 256, NC), b256, 0, stream>>>(tbuf, S);
    k4a_partial<<<NB4, b256, 0, stream>>>(sm, spmap, spi, i, part);
    k4b_finish<<<1, 64, 0, stream>>>(part, B12);
    k5_update<<<(NPIX / 16) / 8, b256, 0, stream>>>(S, sm, ut, Mh, B12, spmap,
                                                    spi, i, loww, highw, q);
  }
  k6_output<<<NPIX / 256, b256, 0, stream>>>(q, out);
}
